// MambaBlock_76012331204736
// MI455X (gfx1250) — compile-verified, matrix operand order checked by reading, not run on hardware
//
#include <hip/hip_runtime.h>
#include <hip/hip_bf16.h>

// ---------------- problem constants (from reference) ----------------
#define B_      2
#define S_      1024
#define DM_     1024          // D_MODEL
#define DI_     2048          // D_INNER = 2*D_MODEL
#define DSTATE_ 16
#define DCONV_  4
#define NROW_   (B_ * S_)     // 2048 flattened (b,s) rows
#define XPN_    (2 * DSTATE_ + 1)   // 33

typedef __attribute__((ext_vector_type(16))) _Float16 v16h;
typedef __attribute__((ext_vector_type(8)))  _Float16 v8h;
typedef __attribute__((ext_vector_type(8)))  float    v8f;

union V16 { v16h v; v8h h2[2]; };

// ---------------------------------------------------------------
// Elementwise f32 -> f16 convert
// ---------------------------------------------------------------
__global__ void cvt_f16_kernel(const float* __restrict__ in,
                               _Float16* __restrict__ out, int n) {
  int i = blockIdx.x * 256 + threadIdx.x;
  if (i < n) out[i] = (_Float16)in[i];
}

// ---------------------------------------------------------------
// Transpose (K x N) f32  ->  (N x K) f16   (Wt[n*K + k] = W[k*N + n])
// ---------------------------------------------------------------
__global__ void transpose_f16_kernel(const float* __restrict__ W,
                                     _Float16* __restrict__ Wt,
                                     int K, int N) {
  int o = blockIdx.x * 256 + threadIdx.x;
  if (o >= K * N) return;
  int nn = o / K;
  int kk = o - nn * K;
  Wt[o] = (_Float16)W[(size_t)kk * N + nn];
}

// ---------------------------------------------------------------
// Fragment load helpers for 16-bit WMMA operands.
// Per CDNA5 ISA (16-bit A 16x32): lane<16 holds K[0..7] and K[16..23];
// lane>=16 holds K[8..15] and K[24..31]; each half is one contiguous
// 16-byte chunk -> two v8h global loads per fragment.
// ---------------------------------------------------------------
__device__ __forceinline__ void load_frag(V16& f, const _Float16* base, int k0) {
  f.h2[0] = *(const v8h*)(base + k0);
  f.h2[1] = *(const v8h*)(base + k0 + 16);
}

// ---------------------------------------------------------------
// WMMA GEMM:  C(MxN, f32) = A(MxK, f16 row-major) * B(KxN) with B given
// transposed: Bt (N x K, f16 row-major).
// Block = 128 threads (4 waves). Each wave computes a 16x64 strip:
// one A fragment reused by 4 B fragments -> 4 WMMA per 32-wide K-step.
// Explicit register double buffering: all loads for K-chunk (k+32) are
// issued before the WMMAs on chunk k, so waits before each WMMA are
// partial (loadcnt<=n) instead of a full load->wait->MMA serialization.
// Requires: M,N multiples of 64; K multiple of 64.
// ---------------------------------------------------------------
__global__ void gemm_wmma_f16_kernel(const _Float16* __restrict__ A,
                                     const _Float16* __restrict__ Bt,
                                     float* __restrict__ C,
                                     int M, int N, int K) {
  const int lane  = threadIdx.x & 31;
  const int wave  = threadIdx.x >> 5;
  const int tileM = blockIdx.x * 64 + wave * 16;
  const int tileN = blockIdx.y * 64;

  const int halfSel = (lane >> 4) << 3;         // 0 or 8
  const int l16     = lane & 15;

  const _Float16* aBase = A  + (size_t)(tileM + l16) * K + halfSel;
  const _Float16* b0p   = Bt + (size_t)(tileN + l16) * K + halfSel;
  const _Float16* b1p   = b0p + (size_t)16 * K;
  const _Float16* b2p   = b0p + (size_t)32 * K;
  const _Float16* b3p   = b0p + (size_t)48 * K;

  v8f acc0 = {}, acc1 = {}, acc2 = {}, acc3 = {};

  // ---- preload K-chunk 0 into buffer set A ----
  V16 aA, bA0, bA1, bA2, bA3;
  V16 aB, bB0, bB1, bB2, bB3;
  load_frag(aA,  aBase, 0);
  load_frag(bA0, b0p,   0);
  load_frag(bA1, b1p,   0);
  load_frag(bA2, b2p,   0);
  load_frag(bA3, b3p,   0);

#define WMMA4(af, f0, f1, f2, f3)                                            \
  acc0 = __builtin_amdgcn_wmma_f32_16x16x32_f16(false, (af).v, false,        \
             (f0).v, (short)0, acc0, false, false);                          \
  acc1 = __builtin_amdgcn_wmma_f32_16x16x32_f16(false, (af).v, false,        \
             (f1).v, (short)0, acc1, false, false);                          \
  acc2 = __builtin_amdgcn_wmma_f32_16x16x32_f16(false, (af).v, false,        \
             (f2).v, (short)0, acc2, false, false);                          \
  acc3 = __builtin_amdgcn_wmma_f32_16x16x32_f16(false, (af).v, false,        \
             (f3).v, (short)0, acc3, false, false)

  for (int k0 = 0; k0 < K; k0 += 64) {
    // issue loads for chunk k0+32 into buffer set B (always in range: K%64==0)
    const int k1 = k0 + 32;
    load_frag(aB,  aBase, k1);
    load_frag(bB0, b0p,   k1);
    load_frag(bB1, b1p,   k1);
    load_frag(bB2, b2p,   k1);
    load_frag(bB3, b3p,   k1);
    __builtin_prefetch(aBase + k1 + 32, 0, 3);     // global_prefetch_b8

    // compute on buffer set A (chunk k0) while set-B loads are in flight
    WMMA4(aA, bA0, bA1, bA2, bA3);

    if (k0 + 64 < K) {
      // issue loads for chunk k0+64 back into buffer set A
      const int k2 = k0 + 64;
      load_frag(aA,  aBase, k2);
      load_frag(bA0, b0p,   k2);
      load_frag(bA1, b1p,   k2);
      load_frag(bA2, b2p,   k2);
      load_frag(bA3, b3p,   k2);
      __builtin_prefetch(aBase + k2 + 32, 0, 3);
    }

    // compute on buffer set B (chunk k0+32)
    WMMA4(aB, bB0, bB1, bB2, bB3);
  }
#undef WMMA4

  // C/D layout: VGPR r -> row (tileM + r) for lanes 0..15,
  //                       row (tileM + r + 8) for lanes 16..31; col = lane&15.
  const int rowOff = (lane < 16) ? 0 : 8;
  float* crow = C + (size_t)(tileM + rowOff) * N + tileN + l16;
#pragma unroll
  for (int r = 0; r < 8; ++r) {
    float* cr = crow + (size_t)r * N;
    cr[0]  = acc0[r];
    cr[16] = acc1[r];
    cr[32] = acc2[r];
    cr[48] = acc3[r];
  }
}

// ---------------------------------------------------------------
// Depthwise causal conv (width 4) + bias + SiLU.
// x_in is columns [0, DI) of xz (NROW x 2*DI).  xc: NROW x DI.
// ---------------------------------------------------------------
__global__ void conv_silu_kernel(const float* __restrict__ xz,
                                 const float* __restrict__ conv_w,
                                 const float* __restrict__ conv_b,
                                 float* __restrict__ xc) {
  int idx = blockIdx.x * 256 + threadIdx.x;            // over NROW_*DI_
  if (idx >= NROW_ * DI_) return;
  int d = idx & (DI_ - 1);
  int r = idx >> 11;                                   // /DI_
  int s = r & (S_ - 1);
  float acc = conv_b[d];
#pragma unroll
  for (int j = 0; j < DCONV_; ++j) {
    int ss = s - (DCONV_ - 1) + j;
    if (ss >= 0) {
      acc += xz[(size_t)(r - (DCONV_ - 1) + j) * (2 * DI_) + d]
             * conv_w[d * DCONV_ + j];
    }
  }
  acc = acc / (1.f + __expf(-acc));                    // SiLU
  xc[idx] = acc;
}

// ---------------------------------------------------------------
// xp = xc @ W_xproj   (NROW x DI) * (DI x 33) -> (NROW x 33)
// tiny-N GEMM: one thread per output element.
// ---------------------------------------------------------------
__global__ void xproj_kernel(const float* __restrict__ xc,
                             const float* __restrict__ Wx,
                             float* __restrict__ xp) {
  int idx = blockIdx.x * 256 + threadIdx.x;
  if (idx >= NROW_ * XPN_) return;
  int j = idx % XPN_;
  int r = idx / XPN_;
  const float* xr = xc + (size_t)r * DI_;
  float acc = 0.f;
#pragma unroll 4
  for (int k = 0; k < DI_; ++k) acc += xr[k] * Wx[k * XPN_ + j];
  xp[idx] = acc;
}

// ---------------------------------------------------------------
// Fused selective scan. One thread per (b, d). State h[16] in regs.
// Per step: broadcast xp row (dt|B|C) through LDS, compute
// softplus(delta), dA=exp(delta*A), recurrence, y = h.C + Dp*xc,
// gate with silu(z), emit f16 for the output GEMM.
// Grid: (DI_/256, B_), block 256.
// ---------------------------------------------------------------
__global__ void ssm_scan_kernel(const float* __restrict__ xp,
                                const float* __restrict__ xc,
                                const float* __restrict__ xz,   // z at cols [DI_, 2*DI_)
                                const float* __restrict__ W_dt, // (1, DI_)
                                const float* __restrict__ b_dt,
                                const float* __restrict__ A_log,
                                const float* __restrict__ Dp,
                                _Float16* __restrict__ y_h) {
  __shared__ float bc[XPN_];
  const int d = blockIdx.x * 256 + threadIdx.x;
  const int b = blockIdx.y;

  float Aneg[DSTATE_];
#pragma unroll
  for (int n = 0; n < DSTATE_; ++n) Aneg[n] = -__expf(A_log[n]);

  const float wdt = W_dt[d];
  const float bdt = b_dt[d];
  const float dpv = Dp[d];

  float h[DSTATE_];
#pragma unroll
  for (int n = 0; n < DSTATE_; ++n) h[n] = 0.f;

  for (int s = 0; s < S_; ++s) {
    const int r = b * S_ + s;
    if (threadIdx.x < XPN_) bc[threadIdx.x] = xp[(size_t)r * XPN_ + threadIdx.x];
    __syncthreads();

    const float din   = bc[0] * wdt + bdt;
    const float delta = (din > 20.f) ? din : __logf(1.f + __expf(din));
    const float xcv   = xc[(size_t)r * DI_ + d];
    const float zv    = xz[(size_t)r * (2 * DI_) + DI_ + d];
    const float dx    = delta * xcv;

    float y = 0.f;
#pragma unroll
    for (int n = 0; n < DSTATE_; ++n) {
      const float e = __expf(delta * Aneg[n]);
      h[n] = e * h[n] + dx * bc[1 + n];
      y += h[n] * bc[1 + DSTATE_ + n];
    }
    y += dpv * xcv;
    y *= zv / (1.f + __expf(-zv));                 // * silu(z)
    y_h[(size_t)r * DI_ + d] = (_Float16)y;
    __syncthreads();                               // protect bc[] reuse
  }
}

// ---------------------------------------------------------------
extern "C" void kernel_launch(void* const* d_in, const int* in_sizes, int n_in,
                              void* d_out, int out_size, void* d_ws, size_t ws_size,
                              hipStream_t stream) {
  const float* x      = (const float*)d_in[0];   // (B,S,DM)
  const float* W_in   = (const float*)d_in[1];   // (DM, 2*DI)
  const float* conv_w = (const float*)d_in[2];   // (DI,1,4)
  const float* conv_b = (const float*)d_in[3];   // (DI)
  const float* W_xp   = (const float*)d_in[4];   // (DI, 33)
  const float* W_dt   = (const float*)d_in[5];   // (1, DI)
  const float* b_dt   = (const float*)d_in[6];   // (DI)
  const float* A_log  = (const float*)d_in[7];   // (16)
  const float* Dp     = (const float*)d_in[8];   // (DI)
  const float* W_out  = (const float*)d_in[9];   // (DI, DM)
  float* out = (float*)d_out;                    // (B,S,DM)

  // ---- workspace carve-out (256B aligned) ----
  size_t off = 0;
  auto carve = [&](size_t bytes) -> void* {
    void* p = (char*)d_ws + off;
    off += (bytes + 255) & ~(size_t)255;
    return p;
  };
  _Float16* x_h  = (_Float16*)carve((size_t)NROW_ * DM_ * 2);      // 4 MB
  _Float16* Wi_t = (_Float16*)carve((size_t)(2*DI_) * DM_ * 2);    // 8 MB  (4096 x 1024)
  _Float16* Wo_t = (_Float16*)carve((size_t)DM_ * DI_ * 2);        // 4 MB  (1024 x 2048)
  float*    xz   = (float*)   carve((size_t)NROW_ * 2*DI_ * 4);    // 32 MB
  float*    xc   = (float*)   carve((size_t)NROW_ * DI_ * 4);      // 16 MB
  float*    xp   = (float*)   carve((size_t)NROW_ * XPN_ * 4);     // ~0.3 MB
  _Float16* y_h  = (_Float16*)carve((size_t)NROW_ * DI_ * 2);      // 8 MB

  // 1) precision conversion / weight transposes
  {
    int n = NROW_ * DM_;
    cvt_f16_kernel<<<(n + 255) / 256, 256, 0, stream>>>(x, x_h, n);
  }
  {
    int n = DM_ * 2 * DI_;   // K=1024, N=4096
    transpose_f16_kernel<<<(n + 255) / 256, 256, 0, stream>>>(W_in, Wi_t, DM_, 2 * DI_);
  }
  {
    int n = DI_ * DM_;       // K=2048, N=1024
    transpose_f16_kernel<<<(n + 255) / 256, 256, 0, stream>>>(W_out, Wo_t, DI_, DM_);
  }

  // 2) xz = x @ W_in        (2048 x 4096, K=1024)  -- WMMA
  {
    dim3 grid(NROW_ / 64, (2 * DI_) / 64);
    gemm_wmma_f16_kernel<<<grid, 128, 0, stream>>>(x_h, Wi_t, xz,
                                                   NROW_, 2 * DI_, DM_);
  }

  // 3) depthwise conv + SiLU
  {
    int n = NROW_ * DI_;
    conv_silu_kernel<<<(n + 255) / 256, 256, 0, stream>>>(xz, conv_w, conv_b, xc);
  }

  // 4) xp = xc @ W_xproj    (N=33)
  {
    int n = NROW_ * XPN_;
    xproj_kernel<<<(n + 255) / 256, 256, 0, stream>>>(xc, W_xp, xp);
  }

  // 5) fused selective scan -> gated y (f16)
  {
    dim3 grid(DI_ / 256, B_);
    ssm_scan_kernel<<<grid, 256, 0, stream>>>(xp, xc, xz, W_dt, b_dt,
                                              A_log, Dp, y_h);
  }

  // 6) out = y @ W_out      (2048 x 1024, K=2048)  -- WMMA
  {
    dim3 grid(NROW_ / 64, DM_ / 64);
    gemm_wmma_f16_kernel<<<grid, 128, 0, stream>>>(y_h, Wo_t, out,
                                                   NROW_, DM_, DI_);
  }
  (void)in_sizes; (void)n_in; (void)out_size; (void)ws_size;
}
